// CausalAttention_50611894616067
// MI455X (gfx1250) — compile-verified, matrix operand order checked by reading, not run on hardware
//
#include <hip/hip_runtime.h>
#include <hip/hip_bf16.h>
#include <math.h>

typedef __attribute__((ext_vector_type(16))) _Float16 v16h;
typedef __attribute__((ext_vector_type(8)))  _Float16 v8h;
typedef __attribute__((ext_vector_type(8)))  float    v8f;

constexpr int EMBED  = 1024;
constexpr int HEADS  = 16;
constexpr int HD     = 64;      // head dim
constexpr int BATCH  = 2;
constexpr int SEQ    = 2048;
constexpr int TOKENS = BATCH * SEQ;          // 4096

// ---------------------------------------------------------------------------
// Fragment loader for 16-bit WMMA operands (both A and B use the same per-lane
// pattern on CDNA5): lane L holds row (rowBase + L%16); elements 0..7 are
// columns colBase + (L/16)*8 + 0..7, elements 8..15 are +16 further.
// Two global_load_b128 per fragment.
// ---------------------------------------------------------------------------
__device__ __forceinline__ v16h load_frag(const _Float16* __restrict__ base,
                                          int rowBase, int rowStride, int colBase) {
  const int lane = threadIdx.x & 31;
  const int r16  = lane & 15;
  const int h8   = (lane >> 4) << 3;
  const _Float16* p = base + (size_t)(rowBase + r16) * rowStride + colBase + h8;
  union { v16h v; v8h h[2]; } u;
  u.h[0] = *(const v8h*)(p);
  u.h[1] = *(const v8h*)(p + 16);
  return u.v;
}

__device__ __forceinline__ v8f wmma_f32_f16(v16h a, v16h b, v8f c) {
  return __builtin_amdgcn_wmma_f32_16x16x32_f16(false, a, false, b,
                                                (short)0, c, false, false);
}

// ---------------------------------------------------------------------------
// Stage 1: f32 -> f16 conversion
// ---------------------------------------------------------------------------
__global__ void cvt_f32_to_f16(const float* __restrict__ src,
                               _Float16* __restrict__ dst, int n) {
  for (int i = blockIdx.x * blockDim.x + threadIdx.x; i < n;
       i += gridDim.x * blockDim.x)
    dst[i] = (_Float16)src[i];
}

// ---------------------------------------------------------------------------
// Stage 2: fused QKV projection.  One wave computes a 32-token x 64-feature
// tile of one of {Q,K,V} (2 M-subtiles x 4 N-subtiles = 8 WMMA per k-step,
// 6 fragment loads).  Q,K stored [B,H,S,D] f16, V stored transposed
// [B,H,D,S] f16 for the PV GEMM.
// ---------------------------------------------------------------------------
__global__ void qkv_proj_kernel(const _Float16* __restrict__ xh,
                                const _Float16* __restrict__ Wqh,
                                const _Float16* __restrict__ Wkh,
                                const _Float16* __restrict__ Wvh,
                                const float* __restrict__ bq,
                                const float* __restrict__ bk,
                                const float* __restrict__ bv,
                                _Float16* __restrict__ Qh,
                                _Float16* __restrict__ Kh,
                                _Float16* __restrict__ Vt) {
  const int MT = TOKENS / 32;     // 128 token tiles (32 tokens each)
  const int NG = EMBED / 64;      // 16 groups of 4 n-subtiles
  const int wave = (blockIdx.x * blockDim.x + threadIdx.x) >> 5;
  if (wave >= 3 * MT * NG) return;
  const int sel = wave / (MT * NG);
  const int rem = wave % (MT * NG);
  const int mt  = rem / NG;
  const int ng  = rem % NG;

  const _Float16* W    = (sel == 0) ? Wqh : (sel == 1) ? Wkh : Wvh;
  const float*    bias = (sel == 0) ? bq  : (sel == 1) ? bk  : bv;

  const v8f vzero = {0.f,0.f,0.f,0.f,0.f,0.f,0.f,0.f};
  v8f c[2][4] = {{vzero, vzero, vzero, vzero}, {vzero, vzero, vzero, vzero}};

  const int t0 = mt * 32;
  for (int kb = 0; kb < EMBED; kb += 32) {
    v16h a0 = load_frag(xh, t0,      EMBED, kb);
    v16h a1 = load_frag(xh, t0 + 16, EMBED, kb);
#pragma unroll
    for (int n = 0; n < 4; ++n) {
      v16h bf = load_frag(W, ng * 64 + n * 16, EMBED, kb);
      c[0][n] = wmma_f32_f16(a0, bf, c[0][n]);
      c[1][n] = wmma_f32_f16(a1, bf, c[1][n]);
    }
  }

  const int lane = threadIdx.x & 31;
  const int r16  = lane & 15;
  const int h8   = (lane >> 4) << 3;
#pragma unroll
  for (int mi = 0; mi < 2; ++mi) {
#pragma unroll
    for (int n = 0; n < 4; ++n) {
      const int f  = ng * 64 + n * 16 + r16;    // output feature (C column)
      const float bv_ = bias[f];
      const int hh = f >> 6;
      const int d  = f & 63;
      union { v8f v; float fl[8]; } u; u.v = c[mi][n];
#pragma unroll
      for (int r = 0; r < 8; ++r) {
        const int tok = t0 + mi * 16 + r + h8;  // C row (token)
        const int bb  = tok / SEQ;
        const int s   = tok % SEQ;
        const float val = u.fl[r] + bv_;
        if (sel == 2)
          Vt[(((size_t)bb * HEADS + hh) * HD + d) * SEQ + s] = (_Float16)val;
        else {
          _Float16* dst = (sel == 0) ? Qh : Kh;
          dst[(((size_t)bb * HEADS + hh) * SEQ + s) * HD + d] = (_Float16)val;
        }
      }
    }
  }
}

// ---------------------------------------------------------------------------
// Stage 3: causal flash-attention.  One wave per (b,h, 16-query tile).
// Computes S^T tiles (M=key, N=query) so the softmax'd scores pack directly
// into the B operand of out^T = V^T x P^T with WMMA K=32.
// ---------------------------------------------------------------------------
__global__ void attn_kernel(const _Float16* __restrict__ Qh,
                            const _Float16* __restrict__ Kh,
                            const _Float16* __restrict__ Vt,
                            _Float16* __restrict__ attnh) {
  const int QT   = SEQ / 16;                       // 128 query tiles
  const int wave = (blockIdx.x * blockDim.x + threadIdx.x) >> 5;
  if (wave >= BATCH * HEADS * QT) return;
  const int bh = wave / QT;
  const int q0 = (wave % QT) * 16;

  const _Float16* Qb = Qh + (size_t)bh * SEQ * HD;
  const _Float16* Kb = Kh + (size_t)bh * SEQ * HD;
  const _Float16* Vb = Vt + (size_t)bh * HD * SEQ;

  const int lane = threadIdx.x & 31;
  const int r16  = lane & 15;
  const int h8   = (lane >> 4) << 3;
  const int q    = q0 + r16;                       // this lane's query column

  // Q fragments (B operand of S^T): lane holds Q row (q0+r16), d-octets.
  v16h qf0 = load_frag(Qb, q0, HD, 0);
  v16h qf1 = load_frag(Qb, q0, HD, 32);

  const v8f vzero = {0.f,0.f,0.f,0.f,0.f,0.f,0.f,0.f};
  v8f acc[4] = {vzero, vzero, vzero, vzero};       // out^T, d-tiles 0..3
  float m = -INFINITY, l = 0.f;
  const float scale = 0.125f;                      // 1/sqrt(64)

  const int nkb = (q0 + 15) / 32 + 1;              // causal key-block count
  for (int kbi = 0; kbi < nkb; ++kbi) {
    const int kb = kbi * 32;
    if (kbi + 1 < nkb) {
      __builtin_prefetch(Kb + (size_t)(kb + 32 + r16) * HD, 0, 0);
      __builtin_prefetch(Vb + (size_t)r16 * SEQ + kb + 32, 0, 0);
    }

    // S^T tiles: A = K rows (16 keys x 32 d), B = Q^T; accumulate over d.
    v8f s0 = vzero, s1 = vzero;
    s0 = wmma_f32_f16(load_frag(Kb, kb,      HD, 0 ), qf0, s0);
    s0 = wmma_f32_f16(load_frag(Kb, kb,      HD, 32), qf1, s0);
    s1 = wmma_f32_f16(load_frag(Kb, kb + 16, HD, 0 ), qf0, s1);
    s1 = wmma_f32_f16(load_frag(Kb, kb + 16, HD, 32), qf1, s1);

    union { v8f v; float f[8]; } u0, u1; u0.v = s0; u1.v = s1;
    // causal mask + scale: key index = kb + tile*16 + h8 + r
#pragma unroll
    for (int r = 0; r < 8; ++r) {
      const int k0 = kb + h8 + r;
      const int k1 = kb + 16 + h8 + r;
      u0.f[r] = (k0 <= q) ? u0.f[r] * scale : -INFINITY;
      u1.f[r] = (k1 <= q) ? u1.f[r] * scale : -INFINITY;
    }

    // online softmax (per query; lanes L and L^16 share a query column)
    float mloc = -INFINITY;
#pragma unroll
    for (int r = 0; r < 8; ++r) mloc = fmaxf(mloc, fmaxf(u0.f[r], u1.f[r]));
    mloc = fmaxf(mloc, __shfl_xor(mloc, 16, 32));
    const float mnew  = fmaxf(m, mloc);
    const float alpha = __expf(m - mnew);
#pragma unroll
    for (int dt = 0; dt < 4; ++dt) {
      union { v8f v; float f[8]; } ua; ua.v = acc[dt];
#pragma unroll
      for (int r = 0; r < 8; ++r) ua.f[r] *= alpha;
      acc[dt] = ua.v;
    }

    union { v16h v; _Float16 e[16]; } pf;          // P^T B-fragment (K=32)
    float lloc = 0.f;
#pragma unroll
    for (int r = 0; r < 8; ++r) {
      const float p0 = __expf(u0.f[r] - mnew);
      const float p1 = __expf(u1.f[r] - mnew);
      pf.e[r]     = (_Float16)p0;                  // keys h8+0..7
      pf.e[8 + r] = (_Float16)p1;                  // keys 16+h8+0..7
      lloc += p0 + p1;
    }
    lloc += __shfl_xor(lloc, 16, 32);
    l = l * alpha + lloc;
    m = mnew;

    // out^T += V^T x P^T ; A = V^T rows (d), contiguous in S thanks to Vt.
#pragma unroll
    for (int dt = 0; dt < 4; ++dt)
      acc[dt] = wmma_f32_f16(load_frag(Vb, dt * 16, SEQ, kb), pf.v, acc[dt]);
  }

  const float invl = 1.0f / l;
  const int bb = bh / HEADS;
  const int hh = bh % HEADS;
#pragma unroll
  for (int dt = 0; dt < 4; ++dt) {
    union { v8f v; float f[8]; } ua; ua.v = acc[dt];
#pragma unroll
    for (int r = 0; r < 8; ++r) {
      const int d = dt * 16 + r + h8;              // out^T row = head dim
      attnh[((size_t)bb * SEQ + q) * EMBED + hh * HD + d] =
          (_Float16)(ua.f[r] * invl);
    }
  }
}

// ---------------------------------------------------------------------------
// Stage 4: output projection  out = attn @ Wo^T + bo  (f32 result),
// 32-token x 64-feature tile per wave (same blocking as stage 2).
// ---------------------------------------------------------------------------
__global__ void out_proj_kernel(const _Float16* __restrict__ ah,
                                const _Float16* __restrict__ Woh,
                                const float* __restrict__ bo,
                                float* __restrict__ out) {
  const int MT = TOKENS / 32;     // 128
  const int NG = EMBED / 64;      // 16
  const int wave = (blockIdx.x * blockDim.x + threadIdx.x) >> 5;
  if (wave >= MT * NG) return;
  const int mt = wave / NG;
  const int ng = wave % NG;

  const v8f vzero = {0.f,0.f,0.f,0.f,0.f,0.f,0.f,0.f};
  v8f c[2][4] = {{vzero, vzero, vzero, vzero}, {vzero, vzero, vzero, vzero}};

  const int t0 = mt * 32;
  for (int kb = 0; kb < EMBED; kb += 32) {
    v16h a0 = load_frag(ah, t0,      EMBED, kb);
    v16h a1 = load_frag(ah, t0 + 16, EMBED, kb);
#pragma unroll
    for (int n = 0; n < 4; ++n) {
      v16h bf = load_frag(Woh, ng * 64 + n * 16, EMBED, kb);
      c[0][n] = wmma_f32_f16(a0, bf, c[0][n]);
      c[1][n] = wmma_f32_f16(a1, bf, c[1][n]);
    }
  }

  const int lane = threadIdx.x & 31;
  const int r16  = lane & 15;
  const int h8   = (lane >> 4) << 3;
#pragma unroll
  for (int mi = 0; mi < 2; ++mi) {
#pragma unroll
    for (int n = 0; n < 4; ++n) {
      const int f = ng * 64 + n * 16 + r16;
      const float bv_ = bo[f];
      union { v8f v; float fl[8]; } u; u.v = c[mi][n];
#pragma unroll
      for (int r = 0; r < 8; ++r) {
        const int tok = t0 + mi * 16 + r + h8;
        out[(size_t)tok * EMBED + f] = u.fl[r] + bv_;
      }
    }
  }
}

// ---------------------------------------------------------------------------
// Launcher.  Inputs: 0=x 1=Wq 2=bq 3=Wk 4=bk 5=Wv 6=bv 7=Wo 8=bo
// ---------------------------------------------------------------------------
extern "C" void kernel_launch(void* const* d_in, const int* in_sizes, int n_in,
                              void* d_out, int out_size, void* d_ws, size_t ws_size,
                              hipStream_t stream) {
  const float* x  = (const float*)d_in[0];
  const float* Wq = (const float*)d_in[1];
  const float* bq = (const float*)d_in[2];
  const float* Wk = (const float*)d_in[3];
  const float* bk = (const float*)d_in[4];
  const float* Wv = (const float*)d_in[5];
  const float* bv = (const float*)d_in[6];
  const float* Wo = (const float*)d_in[7];
  const float* bo = (const float*)d_in[8];
  float* out = (float*)d_out;

  // workspace layout (in _Float16 elements), ~48 MB total
  _Float16* ws = (_Float16*)d_ws;
  const size_t NX = (size_t)TOKENS * EMBED;        // 4,194,304
  const size_t NW = (size_t)EMBED * EMBED;         // 1,048,576
  _Float16* xh  = ws;
  _Float16* Wqh = xh  + NX;
  _Float16* Wkh = Wqh + NW;
  _Float16* Wvh = Wkh + NW;
  _Float16* Wvt = Wvh;                              // alias kept for clarity
  _Float16* Woh = Wvh + NW;
  _Float16* Qh  = Woh + NW;
  _Float16* Kh  = Qh  + NX;
  _Float16* Vt  = Kh  + NX;
  _Float16* ah  = Vt  + NX;                        // attention output f16
  (void)Wvt;

  // Stage 1: conversions
  cvt_f32_to_f16<<<4096, 256, 0, stream>>>(x,  xh,  (int)NX);
  cvt_f32_to_f16<<<1024, 256, 0, stream>>>(Wq, Wqh, (int)NW);
  cvt_f32_to_f16<<<1024, 256, 0, stream>>>(Wk, Wkh, (int)NW);
  cvt_f32_to_f16<<<1024, 256, 0, stream>>>(Wv, Wvh, (int)NW);
  cvt_f32_to_f16<<<1024, 256, 0, stream>>>(Wo, Woh, (int)NW);

  // Stage 2: QKV projection — 3 * 128 * 16 = 6144 waves, 4 waves/block
  qkv_proj_kernel<<<1536, 128, 0, stream>>>(xh, Wqh, Wkh, Wvh, bq, bk, bv,
                                            Qh, Kh, Vt);

  // Stage 3: attention — 4096 waves
  attn_kernel<<<1024, 128, 0, stream>>>(Qh, Kh, Vt, ah);

  // Stage 4: output projection — 2048 waves
  out_proj_kernel<<<512, 128, 0, stream>>>(ah, Woh, bo, out);
}
